// MixtralTransformerDecoder_24446953849473
// MI455X (gfx1250) — compile-verified, matrix operand order checked by reading, not run on hardware
//
#include <hip/hip_runtime.h>
#include <hip/hip_bf16.h>
#include <math.h>

// ---------------------------------------------------------------------------
// Mixtral decoder layer for gfx1250 (MI455X).
// bf16 WMMA GEMMs, double-buffered async/TDM LDS staging, gathered top-2 MoE.
// B=2 S=2048 H=1024 NQ=16 NKV=4 HD=64 I=3584 E=8 TOPK=2 WINDOW=512
// ---------------------------------------------------------------------------

typedef __attribute__((ext_vector_type(16))) __bf16 v16bf;
typedef __attribute__((ext_vector_type(8)))  float  v8f;
typedef __attribute__((ext_vector_type(4)))  unsigned int u32x4;
typedef __attribute__((ext_vector_type(8)))  int i32x8;
typedef __attribute__((ext_vector_type(4)))  int i32x4;

#define TOK   4096      // B*S
#define HDIM  1024
#define SEQ   2048
#define NQH   16
#define NKVH  4
#define HD    64
#define IDIM  3584
#define NEXP  8
#define WIN   512
#define ROWP  72        // LDS row pitch in shorts for 64-K tiles (144B, 16B aligned)
#define TPITCH 40       // convt scratch pitch

#if defined(__has_builtin)
#if __has_builtin(__builtin_amdgcn_tensor_load_to_lds)
#define HAVE_TDM 1
#endif
#endif
#ifndef HAVE_TDM
#define HAVE_TDM 0
#endif

__device__ __forceinline__ unsigned short f2bf(float f) {
    unsigned int u = __float_as_uint(f);
    u += 0x7fffu + ((u >> 16) & 1u);            // round-to-nearest-even
    return (unsigned short)(u >> 16);
}
__device__ __forceinline__ unsigned int pack2(float lo, float hi) {
    return (unsigned int)f2bf(lo) | ((unsigned int)f2bf(hi) << 16);
}
__device__ __forceinline__ float bflo(unsigned int u) { return __uint_as_float(u << 16); }
__device__ __forceinline__ float bfhi(unsigned int u) { return __uint_as_float(u & 0xffff0000u); }
__device__ __forceinline__ v8f zero8() {
    v8f z = {0.f,0.f,0.f,0.f,0.f,0.f,0.f,0.f};
    return z;
}

// LDS byte offset of a generic pointer into __shared__.
__device__ __forceinline__ unsigned int lds_off(const void* p) {
    return (unsigned int)(unsigned long long)p;
}

// Async 16-byte global->LDS copy (ASYNCcnt).
__device__ __forceinline__ void async_b128(void* ldst, const void* gsrc) {
    asm volatile("global_load_async_to_lds_b128 %0, %1, off"
                 :: "v"(lds_off(ldst)), "v"(gsrc) : "memory");
}
__device__ __forceinline__ void wait_async0() {
    asm volatile("s_wait_asynccnt 0x0" ::: "memory");
}
// Copy one 128B LDS row segment (64B per call x2 halves handled by caller).
__device__ __forceinline__ void async_row64(void* ldst, const void* gsrc) {
    async_b128(ldst, gsrc);
    async_b128((unsigned short*)ldst + 8, (const unsigned short*)gsrc + 8);
    async_b128((unsigned short*)ldst + 16, (const unsigned short*)gsrc + 16);
    async_b128((unsigned short*)ldst + 24, (const unsigned short*)gsrc + 24);
}

#if HAVE_TDM
// TDM: DMA a [rows x 64] bf16 tile (row stride lda shorts) from global into
// LDS, padding each 128B row by 16B -> the 144B ROWP layout.
__device__ __forceinline__ void tdm_load_tile(unsigned int ldsaddr,
                                              const unsigned short* gaddr,
                                              int lda, int rows) {
    unsigned long long ga = (unsigned long long)gaddr;
    u32x4 g0;
    g0[0] = 1u;                                    // count=1, user descriptor
    g0[1] = ldsaddr;                               // lds_addr
    g0[2] = (unsigned int)ga;                      // global_addr[31:0]
    g0[3] = (unsigned int)((ga >> 32) & 0x1ffffffu) | (2u << 30); // type=2
    i32x8 g1;
    g1[0] = (1 << 16) | (1 << 20) | (4 << 22) | (3 << 25);
    //       data=2B     pad_en     intv=32dw    amount=4dw
    g1[1] = (int)(((unsigned)lda & 0xffffu) << 16);        // tensor_dim0[15:0]
    g1[2] = (int)(((unsigned)lda >> 16) & 0xffffu);        // dim0[31:16], dim1 lo=0
    g1[3] = (int)(16u | (64u << 16));              // dim1=1<<20, tile_dim0=64
    g1[4] = rows;                                  // tile_dim1=rows, tile_dim2=0
    g1[5] = lda;                                   // tensor_dim0_stride[31:0]
    g1[6] = 0;
    g1[7] = 0;
    i32x4 z4; z4[0] = 0; z4[1] = 0; z4[2] = 0; z4[3] = 0;
    i32x8 z8;
#pragma unroll
    for (int i = 0; i < 8; ++i) z8[i] = 0;
    __builtin_amdgcn_tensor_load_to_lds(g0, g1, z4, z4, z8, 0);
}
#endif

// Load a 16(row)x32(K) bf16 fragment from a [row][0..63] tile with ROWP pitch.
// ksub selects K 0-31 or 32-63.  (ISA 16-bit A-matrix layout; B pre-transposed.)
__device__ __forceinline__ v16bf ldfrag(const unsigned short* tile, int r0,
                                        int ksub, int lane) {
    const unsigned int* p =
        (const unsigned int*)(tile + (size_t)(r0 + (lane & 15)) * ROWP)
        + ksub * 16 + ((lane >> 4) << 2);
    union { v16bf v; unsigned int u[8]; } f;
#pragma unroll
    for (int i = 0; i < 4; ++i) f.u[i] = p[i];
#pragma unroll
    for (int i = 0; i < 4; ++i) f.u[4 + i] = p[8 + i];
    return f.v;
}

// ---------------------------------------------------------------------------
// Weight convert + transpose: fp32 src[K,N] (one matrix per blockIdx.z)
// -> bf16 dst[N,K].
// ---------------------------------------------------------------------------
__global__ __launch_bounds__(256) void convt_kernel(
        const float* __restrict__ src, unsigned short* __restrict__ dst,
        int K, int N) {
    __shared__ unsigned short t[64 * TPITCH];
    int tid = threadIdx.x;
    int nbase = blockIdx.x * 64, kbase = blockIdx.y * 32;
    const float* s = src + (size_t)blockIdx.z * K * N;
    unsigned short* d = dst + (size_t)blockIdx.z * N * K;
#pragma unroll
    for (int j = 0; j < 2; ++j) {
        int lin = tid + j * 256;
        int kk = lin >> 4, n4 = (lin & 15) * 4;
        float4 v = *(const float4*)(s + (size_t)(kbase + kk) * N + nbase + n4);
        t[(n4 + 0) * TPITCH + kk] = f2bf(v.x);
        t[(n4 + 1) * TPITCH + kk] = f2bf(v.y);
        t[(n4 + 2) * TPITCH + kk] = f2bf(v.z);
        t[(n4 + 3) * TPITCH + kk] = f2bf(v.w);
    }
    __syncthreads();
    int r = tid >> 2, seg = (tid & 3) * 8;
    *(uint4*)(d + (size_t)(nbase + r) * K + kbase + seg) = *(const uint4*)(t + r * TPITCH + seg);
}

// ---------------------------------------------------------------------------
// RMSNorm: one block per token, fp32 in -> bf16 (packed) out.
// ---------------------------------------------------------------------------
__global__ __launch_bounds__(256) void rmsnorm_kernel(
        const float* __restrict__ x, const float* __restrict__ scale,
        unsigned short* __restrict__ y) {
    int t = blockIdx.x, tid = threadIdx.x;
    const float* xp = x + (size_t)t * HDIM;
    float ss = 0.f;
#pragma unroll
    for (int i = 0; i < 4; ++i) { float v = xp[tid + i * 256]; ss += v * v; }
    __shared__ float red[256];
    red[tid] = ss; __syncthreads();
    for (int s = 128; s > 0; s >>= 1) {
        if (tid < s) red[tid] += red[tid + s];
        __syncthreads();
    }
    float r = rsqrtf(red[0] * (1.f / HDIM) + 1e-5f);
    unsigned int* yp = (unsigned int*)(y + (size_t)t * HDIM);
#pragma unroll
    for (int i = 0; i < 2; ++i) {
        int p = tid + i * 256;          // pair index
        int h = p * 2;
        yp[p] = pack2(xp[h] * r * scale[h], xp[h + 1] * r * scale[h + 1]);
    }
}

// ---------------------------------------------------------------------------
// bf16 WMMA GEMM: C[M,N] = A[M,K] * BT[N,K]^T (+resid), A/BT bf16, C fp32.
// Block tile 128x128, K-tile 64, double-buffered LDS; A via TDM, BT async.
// ---------------------------------------------------------------------------
__global__ __launch_bounds__(256) void gemm_kernel(
        const unsigned short* __restrict__ A, const unsigned short* __restrict__ BT,
        const float* __restrict__ resid, float* __restrict__ C,
        int K, int lda, int ldbt, int ldc) {
    __shared__ unsigned short lds[2 * 256 * ROWP];   // 2 bufs x (128A+128B) rows

    int tid = threadIdx.x, lane = tid & 31, wave = tid >> 5;
    int rowbase = blockIdx.y * 128, nbase = blockIdx.x * 128;
    int wm = (wave >> 1) * 32, wn = (wave & 1) * 64;

    v8f acc[2][4];
#pragma unroll
    for (int mi = 0; mi < 2; ++mi)
#pragma unroll
        for (int ni = 0; ni < 4; ++ni) acc[mi][ni] = zero8();

    int srow = tid >> 1, shalf = (tid & 1) * 32;
    const unsigned short* aptr = A + (size_t)(rowbase + srow) * lda + shalf;
    const unsigned short* bptr = BT + (size_t)(nbase + srow) * ldbt + shalf;

    auto stage = [&](int buf, int kt) {
        unsigned short* base = lds + buf * 256 * ROWP;
#if HAVE_TDM
        if (wave == 0)
            tdm_load_tile(lds_off(base), A + (size_t)rowbase * lda + kt, lda, 128);
#else
        async_row64(base + srow * ROWP + shalf, aptr + kt);
#endif
        async_row64(base + 128 * ROWP + srow * ROWP + shalf, bptr + kt);
    };

    int nsteps = K >> 6;
    stage(0, 0);
    for (int i = 0; i < nsteps; ++i) {
        wait_async0();
#if HAVE_TDM
        __builtin_amdgcn_s_wait_tensorcnt(0);
#endif
        __syncthreads();
        if (i + 1 < nsteps) stage((i + 1) & 1, (i + 1) * 64);
        const unsigned short* bufA = lds + (i & 1) * 256 * ROWP;
        const unsigned short* bufB = bufA + 128 * ROWP;
#pragma unroll
        for (int ks = 0; ks < 2; ++ks) {
            v16bf af[2], bfr[4];
#pragma unroll
            for (int mi = 0; mi < 2; ++mi) af[mi] = ldfrag(bufA, wm + mi * 16, ks, lane);
#pragma unroll
            for (int ni = 0; ni < 4; ++ni) bfr[ni] = ldfrag(bufB, wn + ni * 16, ks, lane);
#pragma unroll
            for (int mi = 0; mi < 2; ++mi)
#pragma unroll
                for (int ni = 0; ni < 4; ++ni)
                    acc[mi][ni] = __builtin_amdgcn_wmma_f32_16x16x32_bf16(
                        false, af[mi], false, bfr[ni], (short)0, acc[mi][ni], false, false);
        }
    }

    if (resid != nullptr) {
#pragma unroll
        for (int mi = 0; mi < 2; ++mi)
#pragma unroll
            for (int ni = 0; ni < 4; ++ni) {
                int n = nbase + wn + ni * 16 + (lane & 15);
                int mb = rowbase + wm + mi * 16 + ((lane >> 4) << 3);
#pragma unroll
                for (int d = 0; d < 8; ++d) {
                    size_t idx = (size_t)(mb + d) * ldc + n;
                    C[idx] = acc[mi][ni][d] + resid[idx];
                }
            }
    } else {
#pragma unroll
        for (int mi = 0; mi < 2; ++mi)
#pragma unroll
            for (int ni = 0; ni < 4; ++ni) {
                int n = nbase + wn + ni * 16 + (lane & 15);
                int mb = rowbase + wm + mi * 16 + ((lane >> 4) << 3);
#pragma unroll
                for (int d = 0; d < 8; ++d)
                    C[(size_t)(mb + d) * ldc + n] = acc[mi][ni][d];
            }
    }
}

// ---------------------------------------------------------------------------
// Interleaved RoPE on q[T,16,64] and k[T,4,64] (fp32, in place).
// ---------------------------------------------------------------------------
__global__ __launch_bounds__(256) void rope_kernel(float* __restrict__ q,
                                                   float* __restrict__ k) {
    int idx = blockIdx.x * 256 + threadIdx.x;   // TOK*20 threads
    int t = idx / 20, h = idx % 20;
    int s = t & (SEQ - 1);
    float* base = (h < NQH) ? (q + ((size_t)t * NQH + h) * HD)
                            : (k + ((size_t)t * NKVH + (h - NQH)) * HD);
    float x[HD];
#pragma unroll
    for (int d = 0; d < HD; ++d) x[d] = base[d];
#pragma unroll
    for (int i = 0; i < HD / 2; ++i) {
        float fr = (float)s * expf(-(float)(2 * i) * (9.210340371976184f / (float)HD));
        float c = cosf(fr), sn = sinf(fr);
        base[2 * i]     = x[2 * i] * c     - x[HD / 2 + i] * sn;
        base[2 * i + 1] = x[2 * i + 1] * c + x[i] * sn;
    }
}

// ---------------------------------------------------------------------------
// Sliding-window causal attention, online softmax, fp32 math, bf16 out.
// ---------------------------------------------------------------------------
__global__ __launch_bounds__(256) void attn_kernel(
        const float* __restrict__ q, const float* __restrict__ k,
        const float* __restrict__ v, unsigned short* __restrict__ out) {
    __shared__ float ldsK[128 * HD];
    __shared__ float ldsV[128 * HD];
    int b = blockIdx.z, n = blockIdx.y, qbase = blockIdx.x * 256;
    int tid = threadIdx.x, qi = qbase + tid, nk = n >> 2;

    float qr[HD], acc[HD];
    const float* qp = q + ((size_t)(b * SEQ + qi) * NQH + n) * HD;
#pragma unroll
    for (int d = 0; d < HD; ++d) { qr[d] = qp[d]; acc[d] = 0.f; }
    float mx = -1e30f, l = 0.f;

    int kfirst = qbase - (WIN - 1); if (kfirst < 0) kfirst = 0;
    kfirst &= ~127;
    int klast = qbase + 255;

    for (int kt = kfirst; kt <= klast; kt += 128) {
        __syncthreads();
#pragma unroll
        for (int j = 0; j < 8; ++j) {
            int lin = tid + j * 256;
            int kk = lin >> 4, d4 = (lin & 15) * 4;
            size_t src = ((size_t)(b * SEQ + kt + kk) * NKVH + nk) * HD + d4;
            *(float4*)&ldsK[kk * HD + d4] = *(const float4*)(k + src);
            *(float4*)&ldsV[kk * HD + d4] = *(const float4*)(v + src);
        }
        __syncthreads();
        for (int j = 0; j < 128; ++j) {
            int kidx = kt + j;
            const float* kr = &ldsK[j * HD];
            float s = 0.f;
#pragma unroll
            for (int d = 0; d < HD; d += 4) {
                float4 kv = *(const float4*)(kr + d);
                s += qr[d] * kv.x + qr[d + 1] * kv.y + qr[d + 2] * kv.z + qr[d + 3] * kv.w;
            }
            bool valid = (kidx <= qi) && (kidx > qi - WIN);
            float sv = valid ? s * 0.125f : -1e9f;
            float mnew = fmaxf(mx, sv);
            float corr = __expf(mx - mnew);
            float p = valid ? __expf(sv - mnew) : 0.f;
            l = l * corr + p;
            const float* vr = &ldsV[j * HD];
#pragma unroll
            for (int d = 0; d < HD; ++d) acc[d] = acc[d] * corr + p * vr[d];
            mx = mnew;
        }
    }
    float inv = 1.f / l;
    unsigned int* op = (unsigned int*)(out + (size_t)(b * SEQ + qi) * HDIM + n * HD);
#pragma unroll
    for (int d = 0; d < HD / 2; ++d)
        op[d] = pack2(acc[2 * d] * inv, acc[2 * d + 1] * inv);
}

// ---------------------------------------------------------------------------
// Router: per-token logits (bf16 hidden state), softmax, top-2, bucket build.
// ---------------------------------------------------------------------------
__global__ __launch_bounds__(256) void router_kernel(
        const unsigned short* __restrict__ hs2, const float* __restrict__ wr,
        float* __restrict__ logits_out, float* __restrict__ topw,
        int* __restrict__ tope, int* __restrict__ tpos,
        int* __restrict__ counts, int* __restrict__ bucket) {
    int token = blockIdx.x * 8 + (threadIdx.x >> 5);
    int lane = threadIdx.x & 31;
    const unsigned int* hp = (const unsigned int*)(hs2 + (size_t)token * HDIM);
    float a[NEXP];
#pragma unroll
    for (int e = 0; e < NEXP; ++e) a[e] = 0.f;
    for (int p = lane; p < HDIM / 2; p += 32) {
        unsigned int u = hp[p];
        float x0 = bflo(u), x1 = bfhi(u);
        const float* rp = wr + (size_t)(2 * p) * NEXP;
        float4 r0 = *(const float4*)rp;
        float4 r1 = *(const float4*)(rp + 4);
        float4 r2 = *(const float4*)(rp + 8);
        float4 r3 = *(const float4*)(rp + 12);
        a[0] += x0 * r0.x + x1 * r2.x; a[1] += x0 * r0.y + x1 * r2.y;
        a[2] += x0 * r0.z + x1 * r2.z; a[3] += x0 * r0.w + x1 * r2.w;
        a[4] += x0 * r1.x + x1 * r3.x; a[5] += x0 * r1.y + x1 * r3.y;
        a[6] += x0 * r1.z + x1 * r3.z; a[7] += x0 * r1.w + x1 * r3.w;
    }
#pragma unroll
    for (int e = 0; e < NEXP; ++e)
        for (int off = 16; off; off >>= 1) a[e] += __shfl_xor(a[e], off);
    if (lane == 0) {
        float mxv = a[0];
#pragma unroll
        for (int e = 1; e < NEXP; ++e) mxv = fmaxf(mxv, a[e]);
        float p[NEXP], sum = 0.f;
#pragma unroll
        for (int e = 0; e < NEXP; ++e) { p[e] = __expf(a[e] - mxv); sum += p[e]; }
        float inv = 1.f / sum;
#pragma unroll
        for (int e = 0; e < NEXP; ++e) {
            logits_out[(size_t)token * NEXP + e] = a[e];   // pre-softmax logits
            p[e] *= inv;
        }
        int e0 = 0;
#pragma unroll
        for (int e = 1; e < NEXP; ++e) if (p[e] > p[e0]) e0 = e;
        int e1 = (e0 == 0) ? 1 : 0;
#pragma unroll
        for (int e = 0; e < NEXP; ++e) if (e != e0 && p[e] > p[e1]) e1 = e;
        float w0 = p[e0], w1 = p[e1], ws = 1.f / (w0 + w1);
        w0 *= ws; w1 *= ws;
        int p0 = atomicAdd(&counts[e0], 1);
        int p1 = atomicAdd(&counts[e1], 1);
        bucket[e0 * TOK + p0] = token;
        bucket[e1 * TOK + p1] = token;
        tope[token * 2] = e0; tope[token * 2 + 1] = e1;
        topw[token * 2] = w0; topw[token * 2 + 1] = w1;
        tpos[token * 2] = p0; tpos[token * 2 + 1] = p1;
    }
}

__global__ void zero_counts_kernel(int* counts) {
    if (threadIdx.x < NEXP) counts[threadIdx.x] = 0;
}
__global__ void prefix_kernel(const int* counts, int* offsets) {
    if (threadIdx.x == 0) {
        int s = 0;
        for (int e = 0; e < NEXP; ++e) { offsets[e] = s; s += counts[e]; }
        offsets[NEXP] = s;
    }
}

// ---------------------------------------------------------------------------
// MoE gate+up: gathered bf16 rows, fused gate/up WMMA, silu*up -> bf16 act.
// Block tile 128(rows) x 64(I cols); K-tile 64, double-buffered async staging.
// ---------------------------------------------------------------------------
__global__ __launch_bounds__(256) void moe_gateup_kernel(
        const unsigned short* __restrict__ hs2, const unsigned short* __restrict__ wgT,
        const unsigned short* __restrict__ wuT, const int* __restrict__ bucket,
        const int* __restrict__ counts, const int* __restrict__ offsets,
        unsigned short* __restrict__ act) {
    int e = blockIdx.z;
    int cnt = counts[e];
    int rowbase = blockIdx.y * 128;
    if (rowbase >= cnt) return;
    int nbase = blockIdx.x * 64;

    const unsigned short* BgT = wgT + (size_t)e * HDIM * IDIM;
    const unsigned short* BuT = wuT + (size_t)e * HDIM * IDIM;

    __shared__ unsigned short lds[2 * 256 * ROWP];   // bufs: 128A + 64G + 64U rows

    int tid = threadIdx.x, lane = tid & 31, wave = tid >> 5;
    int arow = tid >> 1, ahalf = (tid & 1) * 32;
    int rglob = rowbase + arow;
    int tok = bucket[e * TOK + (rglob < cnt ? rglob : cnt - 1)];
    const unsigned short* aptr = hs2 + (size_t)tok * HDIM + ahalf;
    int brow = tid >> 2, bq = (tid & 3) * 16;
    const unsigned short* gptr = BgT + (size_t)(nbase + brow) * HDIM + bq;
    const unsigned short* uptr = BuT + (size_t)(nbase + brow) * HDIM + bq;

    auto stage = [&](int buf, int kt) {
        unsigned short* base = lds + buf * 256 * ROWP;
        async_row64(base + arow * ROWP + ahalf, aptr + kt);
        async_b128(base + 128 * ROWP + brow * ROWP + bq, gptr + kt);
        async_b128(base + 128 * ROWP + brow * ROWP + bq + 8, gptr + kt + 8);
        async_b128(base + 192 * ROWP + brow * ROWP + bq, uptr + kt);
        async_b128(base + 192 * ROWP + brow * ROWP + bq + 8, uptr + kt + 8);
    };

    v8f gacc[4], uacc[4];
#pragma unroll
    for (int ni = 0; ni < 4; ++ni) { gacc[ni] = zero8(); uacc[ni] = zero8(); }

    int nsteps = HDIM >> 6;
    stage(0, 0);
    for (int i = 0; i < nsteps; ++i) {
        wait_async0();
        __syncthreads();
        if (i + 1 < nsteps) stage((i + 1) & 1, (i + 1) * 64);
        const unsigned short* bufA = lds + (i & 1) * 256 * ROWP;
        const unsigned short* bufG = bufA + 128 * ROWP;
        const unsigned short* bufU = bufA + 192 * ROWP;
#pragma unroll
        for (int ks = 0; ks < 2; ++ks) {
            v16bf af = ldfrag(bufA, wave * 16, ks, lane);
#pragma unroll
            for (int ni = 0; ni < 4; ++ni) {
                v16bf bg = ldfrag(bufG, ni * 16, ks, lane);
                v16bf bu = ldfrag(bufU, ni * 16, ks, lane);
                gacc[ni] = __builtin_amdgcn_wmma_f32_16x16x32_bf16(
                    false, af, false, bg, (short)0, gacc[ni], false, false);
                uacc[ni] = __builtin_amdgcn_wmma_f32_16x16x32_bf16(
                    false, af, false, bu, (short)0, uacc[ni], false, false);
            }
        }
    }

    int obase = offsets[e];
#pragma unroll
    for (int ni = 0; ni < 4; ++ni) {
        int col = nbase + ni * 16 + (lane & 15);
        int mb = rowbase + wave * 16 + ((lane >> 4) << 3);
#pragma unroll
        for (int d = 0; d < 8; ++d) {
            int m = mb + d;
            if (m < cnt) {
                float g = gacc[ni][d], u = uacc[ni][d];
                float sg = g / (1.f + __expf(-g));        // silu
                act[(size_t)(obase + m) * IDIM + col] = f2bf(sg * u);
            }
        }
    }
}

// ---------------------------------------------------------------------------
// MoE down-proj: act(bf16, compacted) x w_downT[e] -> down_out fp32.
// ---------------------------------------------------------------------------
__global__ __launch_bounds__(256) void moe_down_kernel(
        const unsigned short* __restrict__ act, const unsigned short* __restrict__ wdT,
        const int* __restrict__ counts, const int* __restrict__ offsets,
        float* __restrict__ down_out) {
    int e = blockIdx.z;
    int cnt = counts[e];
    int rowbase = blockIdx.y * 128;
    if (rowbase >= cnt) return;
    int nbase = blockIdx.x * 128;
    int obase = offsets[e];
    const unsigned short* BdT = wdT + (size_t)e * IDIM * HDIM;

    __shared__ unsigned short lds[2 * 256 * ROWP];

    int tid = threadIdx.x, lane = tid & 31, wave = tid >> 5;
    int wm = (wave >> 1) * 32, wn = (wave & 1) * 64;

    int srow = tid >> 1, shalf = (tid & 1) * 32;
    int rglob = rowbase + srow;
    int srcrow = obase + (rglob < cnt ? rglob : cnt - 1);
    const unsigned short* aptr = act + (size_t)srcrow * IDIM + shalf;
    const unsigned short* bptr = BdT + (size_t)(nbase + srow) * IDIM + shalf;

    auto stage = [&](int buf, int kt) {
        unsigned short* base = lds + buf * 256 * ROWP;
        async_row64(base + srow * ROWP + shalf, aptr + kt);
        async_row64(base + 128 * ROWP + srow * ROWP + shalf, bptr + kt);
    };

    v8f acc[2][4];
#pragma unroll
    for (int mi = 0; mi < 2; ++mi)
#pragma unroll
        for (int ni = 0; ni < 4; ++ni) acc[mi][ni] = zero8();

    int nsteps = IDIM >> 6;
    stage(0, 0);
    for (int i = 0; i < nsteps; ++i) {
        wait_async0();
        __syncthreads();
        if (i + 1 < nsteps) stage((i + 1) & 1, (i + 1) * 64);
        const unsigned short* bufA = lds + (i & 1) * 256 * ROWP;
        const unsigned short* bufB = bufA + 128 * ROWP;
#pragma unroll
        for (int ks = 0; ks < 2; ++ks) {
            v16bf af[2], bfr[4];
#pragma unroll
            for (int mi = 0; mi < 2; ++mi) af[mi] = ldfrag(bufA, wm + mi * 16, ks, lane);
#pragma unroll
            for (int ni = 0; ni < 4; ++ni) bfr[ni] = ldfrag(bufB, wn + ni * 16, ks, lane);
#pragma unroll
            for (int mi = 0; mi < 2; ++mi)
#pragma unroll
                for (int ni = 0; ni < 4; ++ni)
                    acc[mi][ni] = __builtin_amdgcn_wmma_f32_16x16x32_bf16(
                        false, af[mi], false, bfr[ni], (short)0, acc[mi][ni], false, false);
        }
    }

#pragma unroll
    for (int mi = 0; mi < 2; ++mi)
#pragma unroll
        for (int ni = 0; ni < 4; ++ni) {
            int n = nbase + wn + ni * 16 + (lane & 15);
            int mb = rowbase + wm + mi * 16 + ((lane >> 4) << 3);
#pragma unroll
            for (int d = 0; d < 8; ++d) {
                int m = mb + d;
                if (m < cnt)
                    down_out[(size_t)(obase + m) * HDIM + n] = acc[mi][ni][d];
            }
        }
}

// ---------------------------------------------------------------------------
// Final combine: out = x2 + w0*down(e0 row) + w1*down(e1 row).
// ---------------------------------------------------------------------------
__global__ __launch_bounds__(256) void final_kernel(
        const float* __restrict__ x2, const float* __restrict__ down_out,
        const float* __restrict__ topw, const int* __restrict__ tope,
        const int* __restrict__ tpos, const int* __restrict__ offsets,
        float* __restrict__ out) {
    int t = blockIdx.x, tid = threadIdx.x;
    int e0 = tope[t * 2], e1 = tope[t * 2 + 1];
    float w0 = topw[t * 2], w1 = topw[t * 2 + 1];
    size_t r0 = (size_t)(offsets[e0] + tpos[t * 2]) * HDIM;
    size_t r1 = (size_t)(offsets[e1] + tpos[t * 2 + 1]) * HDIM;
#pragma unroll
    for (int i = 0; i < 4; ++i) {
        int h = tid + i * 256;
        size_t idx = (size_t)t * HDIM + h;
        out[idx] = x2[idx] + w0 * down_out[r0 + h] + w1 * down_out[r1 + h];
    }
}

// ---------------------------------------------------------------------------
extern "C" void kernel_launch(void* const* d_in, const int* in_sizes, int n_in,
                              void* d_out, int out_size, void* d_ws, size_t ws_size,
                              hipStream_t stream) {
    (void)in_sizes; (void)n_in; (void)out_size; (void)ws_size;
    const float* x        = (const float*)d_in[0];
    const float* norm1    = (const float*)d_in[1];
    const float* norm2    = (const float*)d_in[2];
    const float* wq       = (const float*)d_in[3];
    const float* wk       = (const float*)d_in[4];
    const float* wv       = (const float*)d_in[5];
    const float* wo       = (const float*)d_in[6];
    const float* w_router = (const float*)d_in[7];
    const float* w_gate   = (const float*)d_in[8];
    const float* w_up     = (const float*)d_in[9];
    const float* w_down   = (const float*)d_in[10];
    float* out = (float*)d_out;

    char* wsb = (char*)d_ws;
    size_t off = 0;
    auto alloc = [&](size_t bytes) { char* p = wsb + off; off += (bytes + 255) & ~(size_t)255; return p; };
    typedef unsigned short us;
    us*    hs1      = (us*)alloc((size_t)TOK * HDIM * 2);
    float* qb       = (float*)alloc((size_t)TOK * NQH * HD * 4);
    float* kb       = (float*)alloc((size_t)TOK * NKVH * HD * 4);
    float* vb       = (float*)alloc((size_t)TOK * NKVH * HD * 4);
    us*    attn     = (us*)alloc((size_t)TOK * HDIM * 2);
    float* x2       = (float*)alloc((size_t)TOK * HDIM * 4);
    us*    hs2      = (us*)alloc((size_t)TOK * HDIM * 2);
    int*   counts   = (int*)alloc(256);
    int*   offsets  = (int*)alloc(256);
    int*   bucket   = (int*)alloc((size_t)NEXP * TOK * 4);
    int*   tope     = (int*)alloc((size_t)TOK * 2 * 4);
    float* topw     = (float*)alloc((size_t)TOK * 2 * 4);
    int*   tpos     = (int*)alloc((size_t)TOK * 2 * 4);
    us*    act      = (us*)alloc((size_t)2 * TOK * IDIM * 2);
    float* down_out = (float*)alloc((size_t)2 * TOK * HDIM * 4);
    us*    wqT      = (us*)alloc((size_t)HDIM * HDIM * 2);
    us*    wkT      = (us*)alloc((size_t)HDIM * 256 * 2);
    us*    wvT      = (us*)alloc((size_t)HDIM * 256 * 2);
    us*    woT      = (us*)alloc((size_t)HDIM * HDIM * 2);
    us*    wgT      = (us*)alloc((size_t)NEXP * HDIM * IDIM * 2);
    us*    wuT      = (us*)alloc((size_t)NEXP * HDIM * IDIM * 2);
    us*    wdT      = (us*)alloc((size_t)NEXP * IDIM * HDIM * 2);

    // --- weight convert + transpose to bf16 [N,K] ---
    convt_kernel<<<dim3(16, 32, 1), 256, 0, stream>>>(wq, wqT, HDIM, HDIM);
    convt_kernel<<<dim3(4, 32, 1), 256, 0, stream>>>(wk, wkT, HDIM, 256);
    convt_kernel<<<dim3(4, 32, 1), 256, 0, stream>>>(wv, wvT, HDIM, 256);
    convt_kernel<<<dim3(16, 32, 1), 256, 0, stream>>>(wo, woT, HDIM, HDIM);
    convt_kernel<<<dim3(IDIM / 64, 32, NEXP), 256, 0, stream>>>(w_gate, wgT, HDIM, IDIM);
    convt_kernel<<<dim3(IDIM / 64, 32, NEXP), 256, 0, stream>>>(w_up, wuT, HDIM, IDIM);
    convt_kernel<<<dim3(16, IDIM / 32, NEXP), 256, 0, stream>>>(w_down, wdT, IDIM, HDIM);

    // --- attention block ---
    rmsnorm_kernel<<<TOK, 256, 0, stream>>>(x, norm1, hs1);
    gemm_kernel<<<dim3(8, 32), 256, 0, stream>>>(hs1, wqT, nullptr, qb, HDIM, HDIM, HDIM, 1024);
    gemm_kernel<<<dim3(2, 32), 256, 0, stream>>>(hs1, wkT, nullptr, kb, HDIM, HDIM, HDIM, 256);
    gemm_kernel<<<dim3(2, 32), 256, 0, stream>>>(hs1, wvT, nullptr, vb, HDIM, HDIM, HDIM, 256);
    rope_kernel<<<(TOK * 20) / 256, 256, 0, stream>>>(qb, kb);
    attn_kernel<<<dim3(SEQ / 256, NQH, 2), 256, 0, stream>>>(qb, kb, vb, attn);
    gemm_kernel<<<dim3(8, 32), 256, 0, stream>>>(attn, woT, x, x2, HDIM, HDIM, HDIM, HDIM);

    // --- MoE block ---
    rmsnorm_kernel<<<TOK, 256, 0, stream>>>(x2, norm2, hs2);
    zero_counts_kernel<<<1, 32, 0, stream>>>(counts);
    router_kernel<<<TOK / 8, 256, 0, stream>>>(hs2, w_router, out + (size_t)TOK * HDIM,
                                               topw, tope, tpos, counts, bucket);
    prefix_kernel<<<1, 32, 0, stream>>>(counts, offsets);
    moe_gateup_kernel<<<dim3(IDIM / 64, TOK / 128, NEXP), 256, 0, stream>>>(
        hs2, wgT, wuT, bucket, counts, offsets, act);
    moe_down_kernel<<<dim3(HDIM / 128, TOK / 128, NEXP), 256, 0, stream>>>(
        act, wdT, counts, offsets, down_out);
    final_kernel<<<TOK, 256, 0, stream>>>(x2, down_out, topw, tope, tpos, offsets, out);
}
